// TransformerBlock_43525198577896
// MI455X (gfx1250) — matrix operand order repaired and checked by reading, not compiled, not run on hardware
//
#include <hip/hip_runtime.h>
#include <hip/hip_bf16.h>

typedef __bf16 bf16;
typedef __attribute__((ext_vector_type(16))) __bf16 v16bf;
typedef __attribute__((ext_vector_type(8)))  float  v8f;

constexpr int Bc = 2, Sc = 2048, Dc = 1024, Hc = 16, HDc = 64, DMc = 4;
constexpr int BS  = Bc * Sc;    // 4096 rows
constexpr int DFF = DMc * Dc;   // 4096 ffn dim

union ABf32 { v16bf v; uint4 q[2]; };

// A-operand: two 16B chunks per lane (elements 0..7 from p0, 8..15 from p1; k order is set by the caller)
__device__ inline v16bf load_frag_2x16B(const bf16* p0, const bf16* p1) {
  ABf32 u;
  u.q[0] = *reinterpret_cast<const uint4*>(p0);
  u.q[1] = *reinterpret_cast<const uint4*>(p1);
  return u.v;
}
// B-operand: one 32B chunk per lane (column = lane%16, K = 16*half..+15)
__device__ inline v16bf load_frag_32B(const bf16* p) {
  return *reinterpret_cast<const v16bf*>(p);
}

// 16B-per-lane async copy global -> LDS (ASYNCcnt-tracked DMA path).
// VGLOBAL async form: global_load_async_to_lds_b128 vLDSADDR, v[GADDR], off
// LDS operand is the wave-relative LDS byte offset (low 32 bits of the generic
// shared pointer, per ISA 10.2 aperture mapping).
__device__ inline void copy16_async(const bf16* g, bf16* l) {
  const unsigned lofs = (unsigned)(unsigned long long)l;
  asm volatile("global_load_async_to_lds_b128 %0, %1, off"
               :: "v"(lofs), "v"(g)
               : "memory");
}
__device__ inline void wait_async0() {
  asm volatile("s_wait_asynccnt 0x0" ::: "memory");
}

// ---------------------------------------------------------------- convert
__global__ void f32_to_bf16_kernel(const float* __restrict__ in,
                                   bf16* __restrict__ out, size_t n) {
  size_t i = (size_t)blockIdx.x * blockDim.x + threadIdx.x;
  const size_t stride = (size_t)gridDim.x * blockDim.x;
  for (; i < n; i += stride) out[i] = (bf16)in[i];
}

// ---------------------------------------------------------------- layernorm
__global__ __launch_bounds__(256)
void layernorm_bf16(const float* __restrict__ x, const float* __restrict__ scale,
                    const float* __restrict__ shift, bf16* __restrict__ out) {
  const int row = blockIdx.x;
  const int tid = threadIdx.x;                 // 256 threads, 4 floats each
  const float4 xv = reinterpret_cast<const float4*>(x + (size_t)row * Dc)[tid];
  float sum = xv.x + xv.y + xv.z + xv.w;
  float sq  = xv.x*xv.x + xv.y*xv.y + xv.z*xv.z + xv.w*xv.w;
#pragma unroll
  for (int d = 1; d < 32; d <<= 1) {
    sum += __shfl_xor(sum, d, 32);
    sq  += __shfl_xor(sq,  d, 32);
  }
  __shared__ float ssum[8], ssq[8];
  if ((tid & 31) == 0) { ssum[tid >> 5] = sum; ssq[tid >> 5] = sq; }
  __syncthreads();
  float ts = 0.f, tq = 0.f;
#pragma unroll
  for (int w = 0; w < 8; ++w) { ts += ssum[w]; tq += ssq[w]; }
  const float mean = ts * (1.0f / Dc);
  const float var  = tq * (1.0f / Dc) - mean * mean;
  const float rstd = rsqrtf(var + 1e-5f);
  const float vals[4] = {xv.x, xv.y, xv.z, xv.w};
  const int base = tid * 4;
#pragma unroll
  for (int e = 0; e < 4; ++e) {
    const int c = base + e;
    out[(size_t)row * Dc + c] = (bf16)((vals[e] - mean) * rstd * scale[c] + shift[c]);
  }
}

// ---------------------------------------------------------------- GEMM
// y[r,c] = sum_k A[r,k] * W[c,k]   (torch Linear: x @ W^T), K%32==0, M%128==0, N%64==0
// Block: 256 threads = 8 waves, tile 128(M) x 64(N). B tile (64x32 bf16) staged
// in LDS by the async load-to-LDS DMA, double-buffered, rows padded to 40 elems
// (20-dword stride -> conflict-free).
// OUTMODE: 0 = fp32 row-major, 1 = bf16 row-major,
//          2 = bf16 scatter [b,h,s,hd] (Q/K), 3 = bf16 scatter [b,h,hd,s] (V^T)
template<int OUTMODE, bool RELU, bool BIAS, bool RES>
__global__ __launch_bounds__(256)
void gemm_bf16(const bf16* __restrict__ A, const bf16* __restrict__ W,
               const float* __restrict__ bias, const float* __restrict__ res,
               void* __restrict__ outp, int M, int N, int K) {
  constexpr int LDB = 40;                        // padded LDS row stride (bf16)
  __shared__ alignas(16) bf16 bsh[2][64 * LDB];

  const int tid    = threadIdx.x;
  const int lane   = tid & 31;
  const int wave   = tid >> 5;
  const int lane16 = lane & 15;
  const int half   = lane >> 4;
  const int m0 = blockIdx.y * 128 + wave * 16;
  const int n0 = blockIdx.x * 64;

  // per-thread slice of the B-tile copy: 256 threads x 16B == 64x32 bf16
  const int cn = tid >> 2;                        // 0..63  (n within tile)
  const int cc = (tid & 3) * 8;                   // 0,8,16,24 (k chunk)
  const bf16* wsrc = W + (size_t)(n0 + cn) * K + cc;
  bf16* ldst0 = &bsh[0][cn * LDB + cc];
  bf16* ldst1 = &bsh[1][cn * LDB + cc];

  const bf16* arow = A + (size_t)(m0 + lane16) * K;

  v8f acc[4];
#pragma unroll
  for (int j = 0; j < 4; ++j)
#pragma unroll
    for (int i = 0; i < 8; ++i) acc[j][i] = 0.0f;

  copy16_async(wsrc, ldst0);                      // prologue: buffer 0, k=0
  int buf = 0;
  for (int k = 0; k < K; k += 32) {
    wait_async0();                                // my copy of bsh[buf] done
    __syncthreads();                              // everyone's copy done; other buf free
    if (k + 32 < K) copy16_async(wsrc + k + 32, buf ? ldst0 : ldst1);

    const v16bf a = load_frag_2x16B(arow + k + 16 * half, arow + k + 16 * half + 8);
    const bf16* bbase = bsh[buf];
#pragma unroll
    for (int j = 0; j < 4; ++j) {
      const bf16* bp = bbase + (j * 16 + lane16) * LDB + 16 * half;
      const v16bf b = load_frag_2x16B(bp, bp + 8);
      acc[j] = __builtin_amdgcn_wmma_f32_16x16x32_bf16(
          false, a, false, b, (short)0, acc[j], false, false);
    }
    buf ^= 1;
  }

#pragma unroll
  for (int j = 0; j < 4; ++j) {
    const int c = n0 + j * 16 + lane16;
    const float bv = BIAS ? bias[c] : 0.0f;
#pragma unroll
    for (int i = 0; i < 8; ++i) {
      const int r = m0 + i + 8 * half;
      float v = acc[j][i] + bv;
      if (RELU) v = fmaxf(v, 0.0f);
      if (RES)  v += res[(size_t)r * N + c];
      if constexpr (OUTMODE == 0) {
        ((float*)outp)[(size_t)r * N + c] = v;
      } else if constexpr (OUTMODE == 1) {
        ((bf16*)outp)[(size_t)r * N + c] = (bf16)v;
      } else if constexpr (OUTMODE == 2) {
        const int b = r / Sc, s = r % Sc, h = c >> 6, hd = c & 63;
        ((bf16*)outp)[(((size_t)(b * Hc + h)) * Sc + s) * 64 + hd] = (bf16)v;
      } else {
        const int b = r / Sc, s = r % Sc, h = c >> 6, hd = c & 63;
        ((bf16*)outp)[(((size_t)(b * Hc + h)) * 64 + hd) * Sc + s] = (bf16)v;
      }
    }
  }
}

// ---------------------------------------------------------------- attention
// One wave per 16-query tile of one (b,h). Flash-style online softmax.
// Q,K: [b,h,s,hd] bf16; Vt: [b,h,hd,s] bf16; ctx out: [b*s, D] bf16 row-major.
__global__ __launch_bounds__(32)
void attn_kernel(const bf16* __restrict__ Q, const bf16* __restrict__ Km,
                 const bf16* __restrict__ Vt, bf16* __restrict__ ctx) {
  const int lane   = threadIdx.x & 31;
  const int lane16 = lane & 15;
  const int half   = lane >> 4;
  const int bh     = blockIdx.y;
  const int qbase  = blockIdx.x * 16;

  __shared__ alignas(16) bf16 pbuf[16 * 32];   // P tile staged for A-operand reshape

  const bf16* qrow = Q + ((size_t)bh * Sc + (qbase + lane16)) * HDc;
  const v16bf aq0 = load_frag_2x16B(qrow + 8 * half, qrow + 16 + 8 * half);       // hd 0..31
  const v16bf aq1 = load_frag_2x16B(qrow + 32 + 8 * half, qrow + 48 + 8 * half);  // hd 32..63

  v8f acc[4];
  float m[8], l[8];
#pragma unroll
  for (int j = 0; j < 4; ++j)
#pragma unroll
    for (int i = 0; i < 8; ++i) acc[j][i] = 0.0f;
#pragma unroll
  for (int i = 0; i < 8; ++i) { m[i] = -1e30f; l[i] = 0.0f; }

  const int kmax = qbase + 15;               // last key needed (causal)
  for (int kb = 0; kb <= kmax; kb += 32) {
    v8f s[2];
#pragma unroll
    for (int t = 0; t < 2; ++t) {
      const int kcol = kb + t * 16 + lane16; // this lane's key column
      const bf16* krow = Km + ((size_t)bh * Sc + kcol) * HDc;
      const v16bf bk0 = load_frag_2x16B(krow + 8 * half, krow + 16 + 8 * half);
      const v16bf bk1 = load_frag_2x16B(krow + 32 + 8 * half, krow + 48 + 8 * half);
      v8f st;
#pragma unroll
      for (int i = 0; i < 8; ++i) st[i] = 0.0f;
      st = __builtin_amdgcn_wmma_f32_16x16x32_bf16(false, aq0, false, bk0, (short)0, st, false, false);
      st = __builtin_amdgcn_wmma_f32_16x16x32_bf16(false, aq1, false, bk1, (short)0, st, false, false);
#pragma unroll
      for (int i = 0; i < 8; ++i) {
        const int qr = qbase + i + 8 * half;
        s[t][i] = (kcol <= qr) ? st[i] * 0.125f : -1e30f;  // 1/sqrt(64), causal mask
      }
    }
    // ---- online softmax: row stats live per lane-half, aligned with C layout
    float alpha[8];
#pragma unroll
    for (int i = 0; i < 8; ++i) {
      float mx = fmaxf(s[0][i], s[1][i]);
#pragma unroll
      for (int d = 1; d < 16; d <<= 1) mx = fmaxf(mx, __shfl_xor(mx, d, 32));
      const float mn = fmaxf(m[i], mx);
      alpha[i] = __expf(m[i] - mn);
      m[i] = mn;
    }
#pragma unroll
    for (int i = 0; i < 8; ++i) {
      const float p0 = __expf(s[0][i] - m[i]);
      const float p1 = __expf(s[1][i] - m[i]);
      s[0][i] = p0; s[1][i] = p1;
      float sum = p0 + p1;
#pragma unroll
      for (int d = 1; d < 16; d <<= 1) sum += __shfl_xor(sum, d, 32);
      l[i] = l[i] * alpha[i] + sum;
#pragma unroll
      for (int j = 0; j < 4; ++j) acc[j][i] *= alpha[i];
    }
    // ---- reshape P (C layout) -> A layout via LDS
    __syncthreads();
#pragma unroll
    for (int t = 0; t < 2; ++t)
#pragma unroll
      for (int i = 0; i < 8; ++i)
        pbuf[(i + 8 * half) * 32 + t * 16 + lane16] = (bf16)s[t][i];
    __syncthreads();
    const bf16* prow = pbuf + lane16 * 32;
    const v16bf ap = load_frag_2x16B(prow + 16 * half, prow + 16 * half + 8);
    // ---- PV: B operand from V^T, K = key positions (contiguous)
#pragma unroll
    for (int j = 0; j < 4; ++j) {
      const bf16* vrow = Vt + ((size_t)bh * HDc + j * 16 + lane16) * Sc + kb + 16 * half;
      const v16bf bv = load_frag_32B(vrow);
      acc[j] = __builtin_amdgcn_wmma_f32_16x16x32_bf16(false, ap, false, bv, (short)0, acc[j], false, false);
    }
  }
  // ---- normalize + write ctx as plain [B*S, D] bf16 row-major
  const int b = bh / Hc, h = bh % Hc;
#pragma unroll
  for (int j = 0; j < 4; ++j)
#pragma unroll
    for (int i = 0; i < 8; ++i) {
      const int qr = qbase + i + 8 * half;
      const float v = acc[j][i] / l[i];
      ctx[((size_t)(b * Sc + qr)) * Dc + h * HDc + j * 16 + lane16] = (bf16)v;
    }
}

// ---------------------------------------------------------------- host
extern "C" void kernel_launch(void* const* d_in, const int* in_sizes, int n_in,
                              void* d_out, int out_size, void* d_ws, size_t ws_size,
                              hipStream_t stream) {
  (void)in_sizes; (void)n_in; (void)out_size; (void)ws_size;
  const float* x   = (const float*)d_in[0];
  const float* wq  = (const float*)d_in[1];
  const float* wk  = (const float*)d_in[2];
  const float* wv  = (const float*)d_in[3];
  const float* wo  = (const float*)d_in[4];
  const float* bo  = (const float*)d_in[5];
  const float* l1s = (const float*)d_in[6];
  const float* l1b = (const float*)d_in[7];
  const float* l2s = (const float*)d_in[8];
  const float* l2b = (const float*)d_in[9];
  const float* w1  = (const float*)d_in[10];
  const float* b1  = (const float*)d_in[11];
  const float* w2  = (const float*)d_in[12];
  const float* b2  = (const float*)d_in[13];
  float* out = (float*)d_out;

  char* ws = (char*)d_ws;
  size_t off = 0;
  auto alloc = [&](size_t bytes) -> char* {
    off = (off + 255) & ~size_t(255);
    char* p = ws + off; off += bytes; return p;
  };
  bf16*  wqb  = (bf16*)alloc((size_t)Dc * Dc * 2);
  bf16*  wkb  = (bf16*)alloc((size_t)Dc * Dc * 2);
  bf16*  wvb  = (bf16*)alloc((size_t)Dc * Dc * 2);
  bf16*  wob  = (bf16*)alloc((size_t)Dc * Dc * 2);
  bf16*  w1b  = (bf16*)alloc((size_t)DFF * Dc * 2);
  bf16*  w2b  = (bf16*)alloc((size_t)Dc * DFF * 2);
  bf16*  xln  = (bf16*)alloc((size_t)BS * Dc * 2);
  bf16*  qb   = (bf16*)alloc((size_t)BS * Dc * 2);
  bf16*  kb   = (bf16*)alloc((size_t)BS * Dc * 2);
  bf16*  vtb  = (bf16*)alloc((size_t)BS * Dc * 2);
  bf16*  ctxb = (bf16*)alloc((size_t)BS * Dc * 2);
  float* hbuf = (float*)alloc((size_t)BS * Dc * 4);
  bf16*  y2   = (bf16*)alloc((size_t)BS * Dc * 2);
  bf16*  a1   = (bf16*)alloc((size_t)BS * DFF * 2);

  // 1) weights -> bf16
  f32_to_bf16_kernel<<<512, 256, 0, stream>>>(wq, wqb, (size_t)Dc * Dc);
  f32_to_bf16_kernel<<<512, 256, 0, stream>>>(wk, wkb, (size_t)Dc * Dc);
  f32_to_bf16_kernel<<<512, 256, 0, stream>>>(wv, wvb, (size_t)Dc * Dc);
  f32_to_bf16_kernel<<<512, 256, 0, stream>>>(wo, wob, (size_t)Dc * Dc);
  f32_to_bf16_kernel<<<1024, 256, 0, stream>>>(w1, w1b, (size_t)DFF * Dc);
  f32_to_bf16_kernel<<<1024, 256, 0, stream>>>(w2, w2b, (size_t)Dc * DFF);

  // 2) LN1
  layernorm_bf16<<<BS, 256, 0, stream>>>(x, l1s, l1b, xln);

  // 3) QKV projections (Q/K scatter to [b,h,s,hd], V to [b,h,hd,s])
  {
    dim3 grid(Dc / 64, BS / 128), blk(256);
    gemm_bf16<2, false, false, false><<<grid, blk, 0, stream>>>(
        xln, wqb, nullptr, nullptr, qb, BS, Dc, Dc);
    gemm_bf16<2, false, false, false><<<grid, blk, 0, stream>>>(
        xln, wkb, nullptr, nullptr, kb, BS, Dc, Dc);
    gemm_bf16<3, false, false, false><<<grid, blk, 0, stream>>>(
        xln, wvb, nullptr, nullptr, vtb, BS, Dc, Dc);
  }

  // 4) causal flash attention
  attn_kernel<<<dim3(Sc / 16, Bc * Hc), 32, 0, stream>>>(qb, kb, vtb, ctxb);

  // 5) output projection + bias + residual(x) -> h (fp32)
  gemm_bf16<0, false, true, true><<<dim3(Dc / 64, BS / 128), 256, 0, stream>>>(
      ctxb, wob, bo, x, hbuf, BS, Dc, Dc);

  // 6) LN2
  layernorm_bf16<<<BS, 256, 0, stream>>>(hbuf, l2s, l2b, y2);

  // 7) FFN1: relu(y @ w1^T + b1) -> bf16
  gemm_bf16<1, true, true, false><<<dim3(DFF / 64, BS / 128), 256, 0, stream>>>(
      y2, w1b, b1, nullptr, a1, BS, DFF, Dc);

  // 8) FFN2: a1 @ w2^T + b2 + h -> fp32 out
  gemm_bf16<0, false, true, true><<<dim3(Dc / 64, BS / 128), 256, 0, stream>>>(
      a1, w2b, b2, hbuf, out, BS, Dc, DFF);
}
